// MultiHeadGraphAttention_4217657885156
// MI455X (gfx1250) — matrix operand order repaired and checked by reading, not compiled, not run on hardware
//
#include <hip/hip_runtime.h>

#define NEG_SLOPE 0.2f
#define AGG_CHUNK 128

typedef __attribute__((ext_vector_type(16))) _Float16 v16h;
typedef __attribute__((ext_vector_type(8)))  _Float16 v8h;
typedef __attribute__((ext_vector_type(2)))  _Float16 v2h;
typedef __attribute__((ext_vector_type(8)))  float    v8f;
typedef __attribute__((ext_vector_type(4)))  float    v4f;

static __device__ __forceinline__ v8f wmma_f16(v16h a, v16h b, v8f c) {
  // D = A(16x32 f16) * B(32x16 f16) + C(16x16 f32)
  return __builtin_amdgcn_wmma_f32_16x16x32_f16(
      /*neg_a=*/false, a, /*neg_b=*/false, b,
      /*c_mod=*/(short)0, c, /*reuse_a=*/false, /*reuse_b=*/false);
}

// ---------------------------------------------------------------------------
// Pack W (8,64,64) and merge_kernel (512,64) f32 -> f16 in WMMA B-operand
// register order, so GEMM lanes load one contiguous v16h (32B) per k-chunk.
// B layout (16-bit, 32x16): lane half h holds col lane%16, elem i<8 <-> k=8h+i,
// elem i>=8 <-> k=16+8h+(i-8).
// ---------------------------------------------------------------------------
__global__ void gat_pack(const float* __restrict__ W, const float* __restrict__ M,
                         _Float16* __restrict__ pW, _Float16* __restrict__ pM) {
  const int t = blockIdx.x * blockDim.x + threadIdx.x;
  if (t < 32768) {
    // pW[head(3) | utile(2) | chunk(1) | lane(5) | elem(4)]
    const int elem = t & 15;
    const int lane = (t >> 4) & 31;
    const int c    = (t >> 9) & 1;
    const int ut   = (t >> 10) & 3;
    const int hd   = t >> 12;
    const int k    = c * 32 + ((lane < 16) ? 0 : 8) + (elem < 8 ? elem : elem + 8);
    const int col  = ut * 16 + (lane & 15);
    pW[t] = (_Float16)W[hd * 4096 + k * 64 + col];
  } else if (t < 65536) {
    // pM[ntile(2) | chunk(4) | lane(5) | elem(4)]
    const int q    = t - 32768;
    const int elem = q & 15;
    const int lane = (q >> 4) & 31;
    const int c    = (q >> 9) & 15;
    const int nt   = q >> 13;
    const int k    = c * 32 + ((lane < 16) ? 0 : 8) + (elem < 8 ? elem : elem + 8);
    const int col  = nt * 16 + (lane & 15);
    pM[q] = (_Float16)M[k * 64 + col];
  }
}

// ---------------------------------------------------------------------------
// h = X @ W : (N x 64) * (64 x 512) -> h f16 (N x 512), layout [node][head*64+u].
// Block = 128 threads = 4 waves; wave = one 16-col u-tile of one head; all
// four waves together cover the full 64-col head segment of a 16-row tile.
// Each wave keeps both B k-chunks in registers and sweeps 4 row-tiles.
// C tiles staged in LDS, then stored with coalesced b128 (128B/row).
// ---------------------------------------------------------------------------
__global__ void gat_h_gemm(const float* __restrict__ X, const _Float16* __restrict__ pW,
                           _Float16* __restrict__ hF, int N) {
  __shared__ _Float16 tile[16 * 64];

  const int head  = blockIdx.y;
  const int utile = threadIdx.x >> 5;
  const int lane  = threadIdx.x & 31;
  const int m0    = blockIdx.x * 64;

  const v16h b0 = *(const v16h*)(pW + (((head * 4 + utile) * 2 + 0) * 512 + lane * 16));
  const v16h b1 = *(const v16h*)(pW + (((head * 4 + utile) * 2 + 1) * 512 + lane * 16));

  const int kbase = (lane < 16) ? 0 : 8;  // A layout: two contiguous 8-float runs

  for (int mt = 0; mt < 4; ++mt) {
    const int rt = m0 + mt * 16;
    if (rt >= N) break;                    // uniform across block
    int row = rt + (lane & 15);
    if (row >= N) row = N - 1;             // clamp tail reads

    v16h a0, a1;
    const float* xr = X + (size_t)row * 64 + kbase;
#pragma unroll
    for (int i = 0; i < 8; ++i) { a0[i] = (_Float16)xr[i];      a0[8 + i] = (_Float16)xr[16 + i]; }
#pragma unroll
    for (int i = 0; i < 8; ++i) { a1[i] = (_Float16)xr[32 + i]; a1[8 + i] = (_Float16)xr[48 + i]; }

    v8f acc = {};
    acc = wmma_f16(a0, b0, acc);
    acc = wmma_f16(a1, b1, acc);

    // C layout: VGPR j -> row j (lanes 0-15) / row j+8 (lanes 16-31), col = lane%16.
    // Stage to LDS (no global scatter of b16 stores).
    const int colL = utile * 16 + (lane & 15);
    const int rowL = (lane < 16) ? 0 : 8;
#pragma unroll
    for (int j = 0; j < 8; ++j)
      tile[(rowL + j) * 64 + colL] = (_Float16)acc[j];
    __syncthreads();

    // Cooperative coalesced store: 128 threads x 16B = full 16x64 f16 tile.
    const int tr  = threadIdx.x >> 3;  // tile row 0..15
    const int seg = threadIdx.x & 7;   // 8 x 16B segments per 128B row
    if (rt + tr < N) {
      const v8h vseg = *(const v8h*)(tile + tr * 64 + seg * 8);
      *(v8h*)(hF + (size_t)(rt + tr) * 512 + head * 64 + seg * 8) = vseg;
    }
    __syncthreads();
  }
}

// ---------------------------------------------------------------------------
// alpha_src/dst[n][h] = sum_u h[n][h][u] * a[h][u] / a[h][64+u].
// One thread per (node, head); 4 contiguous v16h loads of h.
// ---------------------------------------------------------------------------
__global__ void gat_alpha(const _Float16* __restrict__ hF, const float* __restrict__ a,
                          float* __restrict__ asrc, float* __restrict__ adst, int N) {
  const int t = blockIdx.x * blockDim.x + threadIdx.x;
  if (t >= N * 8) return;
  const int node = t >> 3, head = t & 7;
  const v16h* hv = (const v16h*)(hF + (size_t)node * 512 + head * 64);
  const float* av = a + head * 128;
  float s0 = 0.f, s1 = 0.f;
#pragma unroll
  for (int blk = 0; blk < 4; ++blk) {
    const v16h x = hv[blk];
#pragma unroll
    for (int i = 0; i < 16; ++i) {
      const float xv = (float)x[i];
      const int u = blk * 16 + i;
      s0 += xv * av[u];
      s1 += xv * av[64 + u];
    }
  }
  asrc[t] = s0;
  adst[t] = s1;
}

// ---------------------------------------------------------------------------
// Per-edge softmax numerators: att = exp(clip(leaky(a_s[src]+a_d[dst]))).
// Denominators via f32 global atomics into att_sum[src][h] (400K addresses;
// ~6.4M atomics spread over many L2 channels -> not a bottleneck).
// ---------------------------------------------------------------------------
__global__ void gat_edge(const int* __restrict__ edges, const float* __restrict__ asrc,
                         const float* __restrict__ adst, float* __restrict__ att,
                         float* __restrict__ attSum, int E) {
  const int e = blockIdx.x * blockDim.x + threadIdx.x;
  if (e >= E) return;
  const int s = edges[2 * e];
  const int d = edges[2 * e + 1];
#pragma unroll
  for (int hh = 0; hh < 8; ++hh) {
    float sc = asrc[s * 8 + hh] + adst[d * 8 + hh];
    sc = sc > 0.f ? sc : NEG_SLOPE * sc;            // leaky_relu
    sc = fminf(2.f, fmaxf(-2.f, sc));               // clip
    const float av = __expf(sc);
    att[(size_t)e * 8 + hh] = av;
    atomicAdd(&attSum[s * 8 + hh], av);
  }
}

// ---------------------------------------------------------------------------
// out[src] += h[dst] * att/att_sum[src].  src is sorted -> register-accumulate
// runs of equal src, flush with f32 atomics only at run boundaries.
// Block = 8 waves, wave = one head; lane covers 2 of 64 u-dims (one coalesced
// 128B f16 line per gather). h (51MB f16) is L2-resident on the 192MB L2.
// Each edge is loaded once; next edge rotates through registers while its h
// row is prefetched (global_prefetch_b8).
// ---------------------------------------------------------------------------
__global__ void gat_aggregate(const int* __restrict__ edges, const float* __restrict__ att,
                              const float* __restrict__ attSum, const _Float16* __restrict__ hF,
                              float* __restrict__ outAgg, int E) {
  const int head = threadIdx.x >> 5;
  const int lane = threadIdx.x & 31;
  const int e0   = blockIdx.x * AGG_CHUNK;
  if (e0 >= E) return;
  int eEnd = e0 + AGG_CHUNK;
  if (eEnd > E) eEnd = E;

  float acc0 = 0.f, acc1 = 0.f;
  int sPrev = -1;
  int sCur = edges[2 * e0];
  int dCur = edges[2 * e0 + 1];

  for (int e = e0; e < eEnd; ++e) {
    int sNext = sCur, dNext = dCur;
    if (e + 1 < eEnd) {
      sNext = edges[2 * (e + 1)];
      dNext = edges[2 * (e + 1) + 1];
      __builtin_prefetch(hF + (size_t)dNext * 512 + head * 64 + lane * 2, 0, 0);
    }

    if (sCur != sPrev) {
      if (sPrev >= 0) {
        float* o = outAgg + (size_t)sPrev * 512 + head * 64 + lane * 2;
        atomicAdd(o, acc0);
        atomicAdd(o + 1, acc1);
      }
      acc0 = acc1 = 0.f;
      sPrev = sCur;
    }

    const float w = att[(size_t)e * 8 + head] / attSum[(size_t)sCur * 8 + head];
    const v2h hv  = *(const v2h*)(hF + (size_t)dCur * 512 + head * 64 + lane * 2);
    acc0 += w * (float)hv[0];
    acc1 += w * (float)hv[1];

    sCur = sNext;
    dCur = dNext;
  }
  if (sPrev >= 0) {
    float* o = outAgg + (size_t)sPrev * 512 + head * 64 + lane * 2;
    atomicAdd(o, acc0);
    atomicAdd(o + 1, acc1);
  }
}

// ---------------------------------------------------------------------------
// y = leaky(out @ merge_kernel) : (N x 512) * (512 x 64). K=512 -> 16 WMMAs.
// Block = 4 waves, wave = one 16-col n-tile (block covers all 64 out cols of
// a 16-row tile). C staged in LDS; leaky + coalesced 256B/row store.
// ---------------------------------------------------------------------------
__global__ void gat_merge(const float* __restrict__ outAgg, const _Float16* __restrict__ pM,
                          float* __restrict__ y, int N) {
  __shared__ float tileO[16 * 64];

  const int ntile = threadIdx.x >> 5;
  const int lane  = threadIdx.x & 31;
  const int rt    = blockIdx.x * 16;
  if (rt >= N) return;
  int row = rt + (lane & 15);
  if (row >= N) row = N - 1;
  const int kbase = (lane < 16) ? 0 : 8;

  v8f acc = {};
  for (int c = 0; c < 16; ++c) {
    v16h a;
    const float* ar = outAgg + (size_t)row * 512 + c * 32 + kbase;
#pragma unroll
    for (int i = 0; i < 8; ++i) { a[i] = (_Float16)ar[i]; a[8 + i] = (_Float16)ar[16 + i]; }
    const v16h b = *(const v16h*)(pM + ((size_t)(ntile * 16 + c) * 512 + lane * 16));
    acc = wmma_f16(a, b, acc);
  }

  // Stage C to LDS.
  const int colL = ntile * 16 + (lane & 15);
  const int rowL = (lane < 16) ? 0 : 8;
#pragma unroll
  for (int j = 0; j < 8; ++j)
    tileO[(rowL + j) * 64 + colL] = acc[j];
  __syncthreads();

  // Cooperative store with final leaky_relu: 128 threads x 32B = 16x64 f32.
  const int tr  = threadIdx.x >> 3;  // row 0..15
  const int seg = threadIdx.x & 7;   // 8 x 32B segments per 256B row
  if (rt + tr < N) {
    const v4f* lp = (const v4f*)(tileO + tr * 64 + seg * 8);
    v4f q0 = lp[0], q1 = lp[1];
#pragma unroll
    for (int i = 0; i < 4; ++i) {
      q0[i] = q0[i] > 0.f ? q0[i] : NEG_SLOPE * q0[i];
      q1[i] = q1[i] > 0.f ? q1[i] : NEG_SLOPE * q1[i];
    }
    v4f* op = (v4f*)(y + (size_t)(rt + tr) * 64 + seg * 8);
    op[0] = q0;
    op[1] = q1;
  }
}

// ---------------------------------------------------------------------------
extern "C" void kernel_launch(void* const* d_in, const int* in_sizes, int n_in,
                              void* d_out, int out_size, void* d_ws, size_t ws_size,
                              hipStream_t stream) {
  const float* X     = (const float*)d_in[0];   // (N,64)
  const int*   edges = (const int*)d_in[1];     // (E,2)
  const float* W     = (const float*)d_in[2];   // (8,64,64)
  const float* a     = (const float*)d_in[3];   // (8,128,1)
  const float* M     = (const float*)d_in[4];   // (512,64)
  float*       y     = (float*)d_out;           // (N,64)

  const int N = in_sizes[0] / 64;
  const int E = in_sizes[1] / 2;

  // Workspace carve-up (aligned 256B). Total ~184 MB for N=50K, E=800K.
  char* ws = (char*)d_ws;
  size_t off = 0;
  auto carve = [&](size_t bytes) -> char* {
    char* p = ws + off;
    off = (off + bytes + 255) & ~(size_t)255;
    return p;
  };
  _Float16* hF     = (_Float16*)carve((size_t)N * 512 * 2);  // h in f16 (L2-resident)
  float*    outAgg = (float*)carve((size_t)N * 512 * 4);     // aggregated messages
  float*    att    = (float*)carve((size_t)E * 8 * 4);       // softmax numerators
  float*    attSum = (float*)carve((size_t)N * 8 * 4);       // softmax denominators
  float*    asrc   = (float*)carve((size_t)N * 8 * 4);
  float*    adst   = (float*)carve((size_t)N * 8 * 4);
  _Float16* pW     = (_Float16*)carve(32768 * 2);            // W in WMMA B-layout
  _Float16* pM     = (_Float16*)carve(32768 * 2);            // merge_kernel B-layout

  // Atomic accumulators must start at zero every call (graph-replay safe).
  hipMemsetAsync(outAgg, 0, (size_t)N * 512 * 4, stream);
  hipMemsetAsync(attSum, 0, (size_t)N * 8 * 4, stream);

  gat_pack<<<65536 / 256, 256, 0, stream>>>(W, M, pW, pM);

  dim3 g1((N + 63) / 64, 8);
  gat_h_gemm<<<g1, 128, 0, stream>>>(X, pW, hF, N);

  gat_alpha<<<(N * 8 + 255) / 256, 256, 0, stream>>>(hF, a, asrc, adst, N);

  gat_edge<<<(E + 255) / 256, 256, 0, stream>>>(edges, asrc, adst, att, attSum, E);

  gat_aggregate<<<(E + AGG_CHUNK - 1) / AGG_CHUNK, 256, 0, stream>>>(edges, att, attSum, hF,
                                                                     outAgg, E);

  gat_merge<<<(N + 15) / 16, 128, 0, stream>>>(outAgg, pM, y, N);
}
